// GLMNB_85839216377961
// MI455X (gfx1250) — compile-verified, matrix operand order checked by reading, not run on hardware
//
#include <hip/hip_runtime.h>
#include <hip/hip_bf16.h>
#include <math.h>

// ---------------------------------------------------------------------------
// GLM negative-binomial loss, fused.
//   t   = X_row . beta            (matvec, 300000x256 f32  -> memory bound)
//   mu  = exp(t); w = mu*s2; A = v*s1          (s1,s2,v,r are SCALARS)
//   term= r*(logA - log(A+w)) + y*(t + logS2 - log(A+w)) + lgtab[y]
//   out = -sum(term)
// Matvec done with v_wmma_f32_16x16x32_f16; beta replicated in all 16
// B-columns so every lane holds 8 of the 16 tile dot-products directly.
// ---------------------------------------------------------------------------

typedef __attribute__((ext_vector_type(16))) _Float16 v16h;
typedef __attribute__((ext_vector_type(8)))  _Float16 v8h;
typedef __attribute__((ext_vector_type(8)))  float    v8f;

#define N_VOXEL 300000
#define BETA_DIM 256
#define N_STUDY 1000
#define GAMMA_DIM 8
#define N_TILES (N_VOXEL / 16)   // 18750, exact

// workspace layout (bytes): [0..7] double accum, [8..51] 11 float params,
// [64..575] 256 x _Float16 beta
// params: 0:r 1:logA 2:logS2 3:A 4:s2 5..10: lgamma table for y=0..5

__global__ __launch_bounds__(256)
void glmnb_setup(const float* __restrict__ Z, const float* __restrict__ gw,
                 const float* __restrict__ theta, const float* __restrict__ beta,
                 float* __restrict__ wsF, double* __restrict__ accum,
                 _Float16* __restrict__ beta16) {
    __shared__ float s1s[256], s2s[256];
    const int tid = threadIdx.x;

    float g[GAMMA_DIM];
#pragma unroll
    for (int j = 0; j < GAMMA_DIM; ++j) g[j] = gw[j];

    float s1 = 0.f, s2 = 0.f;
    for (int s = tid; s < N_STUDY; s += 256) {
        float d = 0.f;
#pragma unroll
        for (int j = 0; j < GAMMA_DIM; ++j) d += Z[s * GAMMA_DIM + j] * g[j];
        float mu = expf(d);
        s1 += mu;
        s2 += mu * mu;
    }
    s1s[tid] = s1; s2s[tid] = s2;
    __syncthreads();
    for (int s = 128; s > 0; s >>= 1) {
        if (tid < s) { s1s[tid] += s1s[tid + s]; s2s[tid] += s2s[tid + s]; }
        __syncthreads();
    }
    if (tid == 0) {
        const float S1 = s1s[0], S2 = s2s[0];
        const float sig = 1.f / (1.f + expf(-theta[0]));
        const float alpha = 100.f * (float)N_STUDY * sig + 1e-8f;
        const float est_alpha = alpha * S2 / (S1 * S1);   // mu_X cancels
        const float v = 1.f / est_alpha;
        const float r = v * S1 * S1 / S2;                 // scalar r
        const float A = v * S1;
        wsF[0] = r;
        wsF[1] = logf(A);
        wsF[2] = logf(S2);
        wsF[3] = A;
        wsF[4] = S2;
        const float lgr = lgammaf(r);
#pragma unroll
        for (int k = 0; k < 6; ++k)
            wsF[5 + k] = lgammaf((float)k + r) - lgammaf((float)(k + 1)) - lgr;
        *accum = 0.0;   // must re-zero every call (graph replays)
    }
    // beta -> f16 (tid covers 0..255 == BETA_DIM)
    beta16[tid] = (_Float16)beta[tid];
}

__global__ __launch_bounds__(256)
void glmnb_main(const float* __restrict__ X, const float* __restrict__ y,
                const float* __restrict__ wsF, const _Float16* __restrict__ beta16,
                double* __restrict__ accum) {
    __shared__ float red[256];
    const int tid  = threadIdx.x;
    const int lane = tid & 31;
    const int hi   = lane >> 4;        // 0: rows 0..7 of tile, 1: rows 8..15
    const int M    = lane & 15;        // A-fragment row within tile
    const int wid        = blockIdx.x * (blockDim.x >> 5) + (tid >> 5);
    const int totalWaves = gridDim.x * (blockDim.x >> 5);

    const float r     = wsF[0];
    const float logA  = wsF[1];
    const float logS2 = wsF[2];
    const float A     = wsF[3];
    const float s2    = wsF[4];

    float acc = 0.f;

    for (int tile = wid; tile < N_TILES; tile += totalWaves) {
        const int rowBase = tile * 16;
        const float* xrow = X + (size_t)(rowBase + M) * BETA_DIM;

        // prefetch next tile this wave will touch
        if (tile + totalWaves < N_TILES)
            __builtin_prefetch(X + (size_t)(rowBase + totalWaves * 16 + M) * BETA_DIM, 0, 0);

        v8f c = {};
#pragma unroll
        for (int k0 = 0; k0 < BETA_DIM; k0 += 32) {
            const int ka = k0 + hi * 8;                  // 32B-aligned float offset
            // A fragment: 16 f16 per lane (ISA 16-bit A 16x32 layout)
            const float4 f0 = *(const float4*)(xrow + ka);
            const float4 f1 = *(const float4*)(xrow + ka + 4);
            const float4 f2 = *(const float4*)(xrow + ka + 16);
            const float4 f3 = *(const float4*)(xrow + ka + 20);
            v16h a;
            a[0]=(_Float16)f0.x; a[1]=(_Float16)f0.y; a[2]=(_Float16)f0.z; a[3]=(_Float16)f0.w;
            a[4]=(_Float16)f1.x; a[5]=(_Float16)f1.y; a[6]=(_Float16)f1.z; a[7]=(_Float16)f1.w;
            a[8]=(_Float16)f2.x; a[9]=(_Float16)f2.y; a[10]=(_Float16)f2.z; a[11]=(_Float16)f2.w;
            a[12]=(_Float16)f3.x; a[13]=(_Float16)f3.y; a[14]=(_Float16)f3.z; a[15]=(_Float16)f3.w;
            // B fragment: beta replicated into every column (same K packing)
            const v8h b0 = *(const v8h*)(beta16 + ka);
            const v8h b1 = *(const v8h*)(beta16 + ka + 16);
            v16h b;
#pragma unroll
            for (int i = 0; i < 8; ++i) { b[i] = b0[i]; b[8 + i] = b1[i]; }
            c = __builtin_amdgcn_wmma_f32_16x16x32_f16(false, a, false, b,
                                                       (short)0, c, false, false);
        }

        // All 16 columns identical -> lane holds rows hi*8+0..7 in c[0..7].
        // Active lanes: 0..7 (rows 0..7) and 16..23 (rows 8..15).
        const int j = lane & 7;
        float t = c[0];
        t = (j == 1) ? c[1] : t;
        t = (j == 2) ? c[2] : t;
        t = (j == 3) ? c[3] : t;
        t = (j == 4) ? c[4] : t;
        t = (j == 5) ? c[5] : t;
        t = (j == 6) ? c[6] : t;
        t = (j == 7) ? c[7] : t;

        if (M < 8) {
            const int row = rowBase + hi * 8 + j;
            const float yv = y[row];
            const float mu = expf(t);
            const float w  = mu * s2;
            const float Lg = logf(A + w);
            int k = (int)yv;
            k = k < 0 ? 0 : (k > 5 ? 5 : k);
            acc += r * (logA - Lg) + yv * (t + logS2 - Lg) + wsF[5 + k];
        }
    }

    red[tid] = acc;
    __syncthreads();
    for (int s = 128; s > 0; s >>= 1) {
        if (tid < s) red[tid] += red[tid + s];
        __syncthreads();
    }
    if (tid == 0) atomicAdd(accum, (double)red[0]);
}

__global__ void glmnb_finish(const double* __restrict__ accum, float* __restrict__ out) {
    if (threadIdx.x == 0 && blockIdx.x == 0) out[0] = -(float)(*accum);
}

extern "C" void kernel_launch(void* const* d_in, const int* in_sizes, int n_in,
                              void* d_out, int out_size, void* d_ws, size_t ws_size,
                              hipStream_t stream) {
    (void)in_sizes; (void)n_in; (void)out_size; (void)ws_size;
    const float* X      = (const float*)d_in[0];
    const float* y      = (const float*)d_in[1];
    const float* Z      = (const float*)d_in[2];
    const float* beta_w = (const float*)d_in[3];
    const float* gamma_w= (const float*)d_in[4];
    const float* theta  = (const float*)d_in[5];
    float* out = (float*)d_out;

    double*   accum  = (double*)d_ws;
    float*    wsF    = (float*)((char*)d_ws + 8);
    _Float16* beta16 = (_Float16*)((char*)d_ws + 64);

    glmnb_setup<<<1, 256, 0, stream>>>(Z, gamma_w, theta, beta_w, wsF, accum, beta16);
    glmnb_main<<<512, 256, 0, stream>>>(X, y, wsF, beta16, accum);
    glmnb_finish<<<1, 64, 0, stream>>>(accum, out);
}
